// GraphTransformerLayer_49976239457038
// MI455X (gfx1250) — matrix operand order repaired and checked by reading, not compiled, not run on hardware
//
#include <hip/hip_runtime.h>
#include <hip/hip_bf16.h>
#include <math.h>

#define N_NODES 2048
#define D_MODEL 256
#define NHEAD 8
#define DK 32
#define N_EDGES 65536
#define H2 512
#define LN_EPS 1e-5f

typedef _Float16 half_t;
typedef __attribute__((ext_vector_type(16))) _Float16 v16h;
typedef __attribute__((ext_vector_type(8)))  float    v8f;

// ---------------------------------------------------------------- utilities
__global__ void k_zero_f32(float* __restrict__ p, int n) {
  int i = blockIdx.x * blockDim.x + threadIdx.x;
  int stride = gridDim.x * blockDim.x;
  for (; i < n; i += stride) p[i] = 0.0f;
}

// transpose-convert weight W (K x Ncols, f32) -> Wt (Ncols x K, f16)
__global__ void k_pack_wt(const float* __restrict__ W, half_t* __restrict__ Wt,
                          int K, int Ncols) {
  int i = blockIdx.x * blockDim.x + threadIdx.x;
  int stride = gridDim.x * blockDim.x;
  int total = K * Ncols;
  for (; i < total; i += stride) {
    int k = i / Ncols;
    int n = i - k * Ncols;
    Wt[(size_t)n * K + k] = (half_t)W[i];
  }
}

// eb = edge_attr @ We + be, scattered into dense S[h][src][dst]
__global__ void k_edge_bias(const float* __restrict__ edge_attr,
                            const float* __restrict__ We,
                            const float* __restrict__ be,
                            const int*   __restrict__ edge_index,
                            float* __restrict__ S) {
  int e = blockIdx.x * blockDim.x + threadIdx.x;
  if (e >= N_EDGES) return;
  int src = edge_index[e];
  int dst = edge_index[N_EDGES + e];
  float f[7];
#pragma unroll
  for (int j = 0; j < 7; ++j) f[j] = edge_attr[e * 7 + j];
#pragma unroll
  for (int hh = 0; hh < NHEAD; ++hh) {
    float acc = be[hh];
#pragma unroll
    for (int j = 0; j < 7; ++j) acc += f[j] * We[j * NHEAD + hh];
    S[((size_t)hh * N_NODES + src) * N_NODES + dst] = acc;
  }
}

// ---------------------------------------------------------------- generic WMMA GEMM
// C(MxNcols) = A(MxK,f32) @ W(KxNcols) + bias, W pre-packed as Wt(Ncols x K, f16).
// One wave computes a 16x64 output strip (4 WMMA tiles sharing one A fragment);
// 8 waves (256 threads) per block.
enum { EPI_F32 = 0, EPI_GELU = 1, EPI_ADD = 2, EPI_HALF_SCALED = 3, EPI_HALF = 4, EPI_HALF_T = 5 };

template <int EPI>
__global__ void k_gemm_wmma(const float* __restrict__ A, const half_t* __restrict__ Bt,
                            const float* __restrict__ bias,
                            void* __restrict__ out, const float* __restrict__ add,
                            int M, int K, int Ncols, float scale) {
  const int lane  = threadIdx.x & 31;
  const int wave  = threadIdx.x >> 5;
  const int lr    = lane & 15;
  const int hf    = lane >> 4;
  const int tn    = blockIdx.x;                 // 64-wide column strip
  const int tm    = blockIdx.y * 8 + wave;      // 16-tall row tile
  const int rowA  = tm * 16 + lr;

  v8f c[4] = {{}, {}, {}, {}};
  for (int kk = 0; kk < K; kk += 32) {
    v16h a;
    const float* ap = A + (size_t)rowA * K + kk + hf * 8;
#pragma unroll
    for (int i = 0; i < 8; ++i) a[i]     = (half_t)ap[i];
#pragma unroll
    for (int i = 0; i < 8; ++i) a[8 + i] = (half_t)ap[16 + i];
#pragma unroll
    for (int s = 0; s < 4; ++s) {
      v16h b;
      const half_t* bp = Bt + (size_t)(tn * 64 + s * 16 + lr) * K + kk + hf * 8;
#pragma unroll
      for (int j = 0; j < 16; ++j) b[j] = bp[(j & 7) + ((j >> 3) << 4)];
      c[s] = __builtin_amdgcn_wmma_f32_16x16x32_f16(false, a, false, b, (short)0, c[s],
                                                    false, false);
    }
  }

#pragma unroll
  for (int s = 0; s < 4; ++s) {
#pragma unroll
    for (int v = 0; v < 8; ++v) {
      const int row = tm * 16 + hf * 8 + v;
      const int col = tn * 64 + s * 16 + lr;
      float val = c[s][v] + bias[col];
      if (EPI == EPI_F32) {
        ((float*)out)[(size_t)row * Ncols + col] = val;
      } else if (EPI == EPI_GELU) {
        ((float*)out)[(size_t)row * Ncols + col] =
            0.5f * val * (1.0f + erff(val * 0.70710678118654752f));
      } else if (EPI == EPI_ADD) {
        ((float*)out)[(size_t)row * Ncols + col] = add[(size_t)row * Ncols + col] + val;
      } else if (EPI == EPI_HALF_SCALED) {
        ((half_t*)out)[(size_t)row * Ncols + col] = (half_t)(val * scale);
      } else if (EPI == EPI_HALF) {
        ((half_t*)out)[(size_t)row * Ncols + col] = (half_t)val;
      } else {  // EPI_HALF_T : out[col][row], leading dim M (for V^T)
        ((half_t*)out)[(size_t)col * M + row] = (half_t)val;
      }
    }
  }
}

// ---------------------------------------------------------------- scores: S += Qs . K^T
// DK==32 -> exactly one WMMA per (16q x 16k) tile per head; bias preloaded as C.
__global__ void k_scores(const half_t* __restrict__ Qh, const half_t* __restrict__ Kh,
                         float* __restrict__ S) {
  const int lane = threadIdx.x & 31;
  const int wave = threadIdx.x >> 5;
  const int lr   = lane & 15;
  const int hf   = lane >> 4;
  const int tk   = blockIdx.x;                // key tile 0..127
  const int tq   = blockIdx.y * 8 + wave;     // query tile 0..127
  const int hh   = blockIdx.z;
  const int qrow = tq * 16 + lr;
  const int mcol = tk * 16 + lr;

  v16h a, b;
  const half_t* ap = Qh + (size_t)qrow * D_MODEL + hh * DK + hf * 8;
#pragma unroll
  for (int i = 0; i < 8; ++i) a[i]     = ap[i];
#pragma unroll
  for (int i = 0; i < 8; ++i) a[8 + i] = ap[16 + i];
  const half_t* bp = Kh + (size_t)mcol * D_MODEL + hh * DK + hf * 8;
#pragma unroll
  for (int j = 0; j < 16; ++j) b[j] = bp[(j & 7) + ((j >> 3) << 4)];

  float* Sp = S + (size_t)hh * N_NODES * N_NODES;
  v8f c;
#pragma unroll
  for (int v = 0; v < 8; ++v)
    c[v] = Sp[(size_t)(tq * 16 + hf * 8 + v) * N_NODES + tk * 16 + lr];
  c = __builtin_amdgcn_wmma_f32_16x16x32_f16(false, a, false, b, (short)0, c, false, false);
#pragma unroll
  for (int v = 0; v < 8; ++v)
    Sp[(size_t)(tq * 16 + hf * 8 + v) * N_NODES + tk * 16 + lr] = c[v];
}

// ---------------------------------------------------------------- row softmax, fp16 out
__global__ void k_softmax(const float* __restrict__ S, half_t* __restrict__ P) {
  const int row = blockIdx.x;          // 0 .. H*N-1  ( = h*N + n )
  const int t   = threadIdx.x;         // 256 threads, 8 vals each
  const float* Sr = S + (size_t)row * N_NODES;
  half_t* Pr = P + (size_t)row * N_NODES;
  __shared__ float red[256];
  float vals[8];
  float m = -3.402823e38f;
#pragma unroll
  for (int i = 0; i < 8; ++i) { vals[i] = Sr[t + i * 256]; m = fmaxf(m, vals[i]); }
  red[t] = m; __syncthreads();
  for (int s = 128; s > 0; s >>= 1) { if (t < s) red[t] = fmaxf(red[t], red[t + s]); __syncthreads(); }
  m = red[0]; __syncthreads();
  float sum = 0.0f;
#pragma unroll
  for (int i = 0; i < 8; ++i) { vals[i] = expf(vals[i] - m); sum += vals[i]; }
  red[t] = sum; __syncthreads();
  for (int s = 128; s > 0; s >>= 1) { if (t < s) red[t] += red[t + s]; __syncthreads(); }
  const float inv = 1.0f / red[0];
#pragma unroll
  for (int i = 0; i < 8; ++i) Pr[t + i * 256] = (half_t)(vals[i] * inv);
}

// ---------------------------------------------------------------- out = P @ V  (per head)
// One wave computes a 16q x 32d strip (both d-tiles of a head) sharing the A fragment.
__global__ void k_attnv(const half_t* __restrict__ P, const half_t* __restrict__ Vt,
                        float* __restrict__ O) {
  const int lane = threadIdx.x & 31;
  const int wave = threadIdx.x >> 5;
  const int lr   = lane & 15;
  const int hf   = lane >> 4;
  const int tq   = blockIdx.x * 8 + wave;      // query tile 0..127
  const int hh   = blockIdx.y;
  const int qrow = tq * 16 + lr;

  v8f c0 = {}, c1 = {};
  const half_t* Pr  = P  + ((size_t)hh * N_NODES + qrow) * N_NODES;
  const half_t* Vr0 = Vt + (size_t)(hh * DK + lr) * N_NODES;
  const half_t* Vr1 = Vt + (size_t)(hh * DK + 16 + lr) * N_NODES;
  for (int mc = 0; mc < N_NODES; mc += 32) {
    v16h a, b0, b1;
    const half_t* ap = Pr + mc + hf * 8;
#pragma unroll
    for (int i = 0; i < 8; ++i) a[i]     = ap[i];
#pragma unroll
    for (int i = 0; i < 8; ++i) a[8 + i] = ap[16 + i];
    const half_t* bp0 = Vr0 + mc + hf * 8;
    const half_t* bp1 = Vr1 + mc + hf * 8;
#pragma unroll
    for (int j = 0; j < 16; ++j) b0[j] = bp0[(j & 7) + ((j >> 3) << 4)];
#pragma unroll
    for (int j = 0; j < 16; ++j) b1[j] = bp1[(j & 7) + ((j >> 3) << 4)];
    c0 = __builtin_amdgcn_wmma_f32_16x16x32_f16(false, a, false, b0, (short)0, c0, false, false);
    c1 = __builtin_amdgcn_wmma_f32_16x16x32_f16(false, a, false, b1, (short)0, c1, false, false);
  }
#pragma unroll
  for (int v = 0; v < 8; ++v) {
    const int row = tq * 16 + hf * 8 + v;
    O[(size_t)row * D_MODEL + hh * DK + lr]      = c0[v];
    O[(size_t)row * D_MODEL + hh * DK + 16 + lr] = c1[v];
  }
}

// ---------------------------------------------------------------- fused LN1 + feature-LN
__global__ void k_ln2(const float* __restrict__ res, const float* __restrict__ O,
                      const float* __restrict__ g1, const float* __restrict__ b1,
                      const float* __restrict__ g2, const float* __restrict__ b2,
                      float* __restrict__ h1, float* __restrict__ x2) {
  const int n = blockIdx.x;
  const int t = threadIdx.x;  // 256 == D_MODEL
  __shared__ float red[256];
  float x = res[(size_t)n * D_MODEL + t] + O[(size_t)n * D_MODEL + t];

  red[t] = x; __syncthreads();
  for (int s = 128; s > 0; s >>= 1) { if (t < s) red[t] += red[t + s]; __syncthreads(); }
  float mu = red[0] * (1.0f / D_MODEL); __syncthreads();
  float d = x - mu;
  red[t] = d * d; __syncthreads();
  for (int s = 128; s > 0; s >>= 1) { if (t < s) red[t] += red[t + s]; __syncthreads(); }
  float var = red[0] * (1.0f / D_MODEL); __syncthreads();
  float h1v = d * rsqrtf(var + LN_EPS) * g1[t] + b1[t];
  h1[(size_t)n * D_MODEL + t] = h1v;

  red[t] = h1v; __syncthreads();
  for (int s = 128; s > 0; s >>= 1) { if (t < s) red[t] += red[t + s]; __syncthreads(); }
  float mu2 = red[0] * (1.0f / D_MODEL); __syncthreads();
  float d2 = h1v - mu2;
  red[t] = d2 * d2; __syncthreads();
  for (int s = 128; s > 0; s >>= 1) { if (t < s) red[t] += red[t + s]; __syncthreads(); }
  float var2 = red[0] * (1.0f / D_MODEL);
  x2[(size_t)n * D_MODEL + t] = d2 * rsqrtf(var2 + LN_EPS) * g2[t] + b2[t];
}

// ---------------------------------------------------------------- launch
extern "C" void kernel_launch(void* const* d_in, const int* in_sizes, int n_in,
                              void* d_out, int out_size, void* d_ws, size_t ws_size,
                              hipStream_t stream) {
  const float* h    = (const float*)d_in[0];
  const float* ea   = (const float*)d_in[1];
  const float* Wq   = (const float*)d_in[2];
  const float* bq   = (const float*)d_in[3];
  const float* Wk   = (const float*)d_in[4];
  const float* bk   = (const float*)d_in[5];
  const float* Wv   = (const float*)d_in[6];
  const float* bv   = (const float*)d_in[7];
  const float* Wo   = (const float*)d_in[8];
  const float* bo   = (const float*)d_in[9];
  const float* We   = (const float*)d_in[10];
  const float* be   = (const float*)d_in[11];
  const float* ln1g = (const float*)d_in[12];
  const float* ln1b = (const float*)d_in[13];
  const float* flng = (const float*)d_in[14];
  const float* flnb = (const float*)d_in[15];
  const float* W1   = (const float*)d_in[16];
  const float* b1   = (const float*)d_in[17];
  const float* W2   = (const float*)d_in[18];
  const float* b2   = (const float*)d_in[19];
  const int*   eidx = (const int*)  d_in[20];
  float* out        = (float*)d_out;

  // workspace carve-out (256B aligned)
  char* w = (char*)d_ws;
  size_t off = 0;
  auto carve = [&](size_t bytes) -> void* {
    void* p = w + off;
    off = (off + bytes + 255) & ~(size_t)255;
    return p;
  };
  const size_t SN = (size_t)NHEAD * N_NODES * N_NODES;
  float*  S    = (float*) carve(SN * 4);                      // 128 MiB scores (+bias)
  half_t* P    = (half_t*)carve(SN * 2);                      //  64 MiB probs (fp16)
  half_t* Qh   = (half_t*)carve((size_t)N_NODES * D_MODEL * 2);
  half_t* Kh   = (half_t*)carve((size_t)N_NODES * D_MODEL * 2);
  half_t* Vt   = (half_t*)carve((size_t)N_NODES * D_MODEL * 2);
  float*  AO   = (float*) carve((size_t)N_NODES * D_MODEL * 4);
  float*  Opj  = (float*) carve((size_t)N_NODES * D_MODEL * 4);
  float*  h1   = (float*) carve((size_t)N_NODES * D_MODEL * 4);
  float*  x2   = (float*) carve((size_t)N_NODES * D_MODEL * 4);
  float*  G    = (float*) carve((size_t)N_NODES * H2 * 4);
  half_t* Wqt  = (half_t*)carve((size_t)D_MODEL * D_MODEL * 2);
  half_t* Wkt  = (half_t*)carve((size_t)D_MODEL * D_MODEL * 2);
  half_t* Wvt  = (half_t*)carve((size_t)D_MODEL * D_MODEL * 2);
  half_t* Wot  = (half_t*)carve((size_t)D_MODEL * D_MODEL * 2);
  half_t* W1t  = (half_t*)carve((size_t)D_MODEL * H2 * 2);
  half_t* W2t  = (half_t*)carve((size_t)H2 * D_MODEL * 2);
  (void)ws_size; (void)n_in; (void)in_sizes; (void)out_size;

  // 0) pack weights: f32 (K x N) -> f16 transposed (N x K)
  k_pack_wt<<<256, 256, 0, stream>>>(Wq, Wqt, D_MODEL, D_MODEL);
  k_pack_wt<<<256, 256, 0, stream>>>(Wk, Wkt, D_MODEL, D_MODEL);
  k_pack_wt<<<256, 256, 0, stream>>>(Wv, Wvt, D_MODEL, D_MODEL);
  k_pack_wt<<<256, 256, 0, stream>>>(Wo, Wot, D_MODEL, D_MODEL);
  k_pack_wt<<<512, 256, 0, stream>>>(W1, W1t, D_MODEL, H2);
  k_pack_wt<<<512, 256, 0, stream>>>(W2, W2t, H2, D_MODEL);

  // 1) dense bias: zero + scatter
  k_zero_f32<<<8192, 256, 0, stream>>>(S, (int)SN);
  k_edge_bias<<<N_EDGES / 256, 256, 0, stream>>>(ea, We, be, eidx, S);

  // 2) QKV projections (fp16 outputs; Q pre-scaled by 1/sqrt(DK); V transposed)
  const dim3 gproj(D_MODEL / 64, N_NODES / 128);
  k_gemm_wmma<EPI_HALF_SCALED><<<gproj, 256, 0, stream>>>(h, Wqt, bq, Qh, nullptr,
      N_NODES, D_MODEL, D_MODEL, 0.17677669529663689f);
  k_gemm_wmma<EPI_HALF><<<gproj, 256, 0, stream>>>(h, Wkt, bk, Kh, nullptr,
      N_NODES, D_MODEL, D_MODEL, 1.0f);
  k_gemm_wmma<EPI_HALF_T><<<gproj, 256, 0, stream>>>(h, Wvt, bv, Vt, nullptr,
      N_NODES, D_MODEL, D_MODEL, 1.0f);

  // 3) scores = Qs.K^T + bias (in place over S)
  k_scores<<<dim3(N_NODES / 16, N_NODES / 128, NHEAD), 256, 0, stream>>>(Qh, Kh, S);

  // 4) softmax over keys -> fp16 P
  k_softmax<<<NHEAD * N_NODES, 256, 0, stream>>>(S, P);

  // 5) attn @ V
  k_attnv<<<dim3(N_NODES / 128, NHEAD), 256, 0, stream>>>(P, Vt, AO);

  // 6) output projection
  k_gemm_wmma<EPI_F32><<<gproj, 256, 0, stream>>>(AO, Wot, bo, Opj, nullptr,
      N_NODES, D_MODEL, D_MODEL, 1.0f);

  // 7) h1 = LN(res + Opj); x2 = LN(h1)
  k_ln2<<<N_NODES, 256, 0, stream>>>(h, Opj, ln1g, ln1b, flng, flnb, h1, x2);

  // 8) FFN: G = gelu(x2@W1+b1); out = h1 + G@W2 + b2
  k_gemm_wmma<EPI_GELU><<<dim3(H2 / 64, N_NODES / 128), 256, 0, stream>>>(x2, W1t, b1, G, nullptr,
      N_NODES, D_MODEL, H2, 1.0f);
  k_gemm_wmma<EPI_ADD><<<gproj, 256, 0, stream>>>(G, W2t, b2, out, h1,
      N_NODES, H2, D_MODEL, 1.0f);
}
